// VisionMamba_71751723647175
// MI455X (gfx1250) — compile-verified, matrix operand order checked by reading, not run on hardware
//
#include <hip/hip_runtime.h>
#include <hip/hip_bf16.h>

typedef __attribute__((ext_vector_type(16))) _Float16 v16h;
typedef __attribute__((ext_vector_type(2)))  _Float16 v2h;
typedef __attribute__((ext_vector_type(8)))  float    v8f;

#define BN_EPS 1e-5f

// ---------------------------------------------------------------------------
// Stem: 7x7 conv stride2 pad3 + bias + BN + ReLU.  x:[4,3,224,224] -> [4,64,112,112]
// ---------------------------------------------------------------------------
__global__ void stem_conv_kernel(const float* __restrict__ x, const float* __restrict__ w,
                                 const float* __restrict__ bias,
                                 const float* __restrict__ g, const float* __restrict__ bb,
                                 const float* __restrict__ mm, const float* __restrict__ vv,
                                 float* __restrict__ y) {
    const int Ho = 112, Wo = 112, Cout = 64, Cin = 3, Hin = 224, Win = 224;
    int idx = blockIdx.x * blockDim.x + threadIdx.x;
    int total = 4 * Cout * Ho * Wo;
    if (idx >= total) return;
    int wo = idx % Wo; int t = idx / Wo;
    int ho = t % Ho;   t /= Ho;
    int co = t % Cout; int b = t / Cout;
    float acc = bias[co];
    const float* wp = w + (size_t)co * Cin * 49;
    const float* ip = x + (size_t)b * Cin * Hin * Win;
    int hi0 = ho * 2 - 3, wi0 = wo * 2 - 3;
    for (int ci = 0; ci < Cin; ++ci) {
        const float* ic = ip + (size_t)ci * Hin * Win;
        const float* wc = wp + ci * 49;
        #pragma unroll
        for (int kh = 0; kh < 7; ++kh) {
            int hi = hi0 + kh; if ((unsigned)hi >= (unsigned)Hin) continue;
            #pragma unroll
            for (int kw = 0; kw < 7; ++kw) {
                int wi = wi0 + kw; if ((unsigned)wi >= (unsigned)Win) continue;
                acc += ic[hi * Win + wi] * wc[kh * 7 + kw];
            }
        }
    }
    float sc = g[co] * rsqrtf(vv[co] + BN_EPS);
    float r  = (acc - mm[co]) * sc + bb[co];
    y[idx] = r > 0.f ? r : 0.f;
}

// ---------------------------------------------------------------------------
// Maxpool 3x3 stride2 pad1: [4,64,112,112] -> [4,64,56,56]
// ---------------------------------------------------------------------------
__global__ void maxpool_kernel(const float* __restrict__ in, float* __restrict__ out) {
    const int C = 64, Hin = 112, Win = 112, Ho = 56, Wo = 56;
    int idx = blockIdx.x * blockDim.x + threadIdx.x;
    int total = 4 * C * Ho * Wo;
    if (idx >= total) return;
    int wo = idx % Wo; int t = idx / Wo;
    int ho = t % Ho;   t /= Ho;
    int c  = t % C;    int b = t / C;
    const float* ip = in + ((size_t)(b * C + c)) * Hin * Win;
    float m = -3.402823466e38f;
    #pragma unroll
    for (int kh = 0; kh < 3; ++kh) {
        int hi = ho * 2 - 1 + kh; if ((unsigned)hi >= (unsigned)Hin) continue;
        #pragma unroll
        for (int kw = 0; kw < 3; ++kw) {
            int wi = wo * 2 - 1 + kw; if ((unsigned)wi >= (unsigned)Win) continue;
            float v = ip[hi * Win + wi];
            m = v > m ? v : m;
        }
    }
    out[idx] = m;
}

// ---------------------------------------------------------------------------
// Stage conv: 3x3 conv (pad1, stride s) + BN + ReLU, fp32 direct.
// ---------------------------------------------------------------------------
__global__ void conv3x3_bn_relu_kernel(const float* __restrict__ in, const float* __restrict__ w,
                                       const float* __restrict__ g, const float* __restrict__ bb,
                                       const float* __restrict__ mm, const float* __restrict__ vv,
                                       float* __restrict__ out,
                                       int Cin, int Hin, int Win, int Cout, int Ho, int Wo, int stride) {
    int idx = blockIdx.x * blockDim.x + threadIdx.x;
    int total = 4 * Cout * Ho * Wo;
    if (idx >= total) return;
    int wo = idx % Wo; int t = idx / Wo;
    int ho = t % Ho;   t /= Ho;
    int co = t % Cout; int b = t / Cout;
    float acc = 0.f;
    const float* wp = w + (size_t)co * Cin * 9;
    const float* ip = in + (size_t)b * Cin * Hin * Win;
    int hi0 = ho * stride - 1, wi0 = wo * stride - 1;
    for (int ci = 0; ci < Cin; ++ci) {
        const float* ic = ip + (size_t)ci * Hin * Win;
        const float* wc = wp + ci * 9;
        #pragma unroll
        for (int kh = 0; kh < 3; ++kh) {
            int hi = hi0 + kh; if ((unsigned)hi >= (unsigned)Hin) continue;
            #pragma unroll
            for (int kw = 0; kw < 3; ++kw) {
                int wi = wi0 + kw; if ((unsigned)wi >= (unsigned)Win) continue;
                acc += ic[hi * Win + wi] * wc[kh * 3 + kw];
            }
        }
    }
    float sc = g[co] * rsqrtf(vv[co] + BN_EPS);
    float r  = (acc - mm[co]) * sc + bb[co];
    out[idx] = r > 0.f ? r : 0.f;
}

// ---------------------------------------------------------------------------
// Layout transposes: NCHW <-> [B, L, C]  (L = H*W)
// ---------------------------------------------------------------------------
__global__ void chw_to_lc_kernel(const float* __restrict__ in, float* __restrict__ out, int C, int L) {
    int idx = blockIdx.x * blockDim.x + threadIdx.x;
    int total = 4 * C * L;
    if (idx >= total) return;
    int c = idx % C; int t2 = idx / C;
    int t = t2 % L;  int b = t2 / L;
    out[idx] = in[((size_t)(b * C + c)) * L + t];
}
__global__ void lc_to_chw_kernel(const float* __restrict__ in, float* __restrict__ out, int C, int L) {
    int idx = blockIdx.x * blockDim.x + threadIdx.x;
    int total = 4 * C * L;
    if (idx >= total) return;
    int t = idx % L; int t2 = idx / L;
    int c = t2 % C;  int b = t2 / C;
    out[idx] = in[((size_t)(b * L + t)) * C + c];
}

// ---------------------------------------------------------------------------
// Packed f32->f16 convert (v_cvt_pk_f16_f32); builtin returns an __fp16
// vector, bit-cast to the _Float16 vector type the WMMA builtin expects.
// ---------------------------------------------------------------------------
static __device__ __forceinline__ v2h pkh(float x, float y) {
    return __builtin_bit_cast(v2h, __builtin_amdgcn_cvt_pkrtz(x, y));
}

// ---------------------------------------------------------------------------
// Fragment load: 16 halves for the CDNA5 16-bit WMMA operand layout.
// p points at (row, kb): halves = p[0..7] and p[16..23].
// Branch-free: four global_load_b128 + eight v_cvt_pk_f16_f32.
// ---------------------------------------------------------------------------
static __device__ __forceinline__ v16h frag_from(const float* __restrict__ p) {
    const float4* q = (const float4*)p;
    float4 a = q[0];    // k +0..3
    float4 b = q[1];    // k +4..7
    float4 c = q[4];    // k +16..19
    float4 d = q[5];    // k +20..23
    v16h r; v2h t;
    t = pkh(a.x, a.y); r[0]  = t[0]; r[1]  = t[1];
    t = pkh(a.z, a.w); r[2]  = t[0]; r[3]  = t[1];
    t = pkh(b.x, b.y); r[4]  = t[0]; r[5]  = t[1];
    t = pkh(b.z, b.w); r[6]  = t[0]; r[7]  = t[1];
    t = pkh(c.x, c.y); r[8]  = t[0]; r[9]  = t[1];
    t = pkh(c.z, c.w); r[10] = t[0]; r[11] = t[1];
    t = pkh(d.x, d.y); r[12] = t[0]; r[13] = t[1];
    t = pkh(d.z, d.w); r[14] = t[0]; r[15] = t[1];
    return r;
}

// ---------------------------------------------------------------------------
// WMMA GEMM: out[M,N] = A[M,K](f32) x W[N,K](f32), f32 accumulate via
// v_wmma_f32_16x16x32_f16.  Each wave computes a 32x32 patch (2x2 tiles of
// 16x16, 4 accumulators) so every A/B fragment feeds two WMMAs.
// Out-of-range rows/cols are handled by CLAMPING the load pointers (a garbage
// A-row m only affects output row m; garbage B-row n only affects column n;
// both are masked at the store) -> the K-loop is completely branch-free.
// K must be a multiple of 32 (true for all projections here).
// ---------------------------------------------------------------------------
__global__ void wmma_gemm_kernel(const float* __restrict__ A, const float* __restrict__ W,
                                 float* __restrict__ C, int M, int N, int K) {
    const int lane = threadIdx.x & 31;
    const int wave = threadIdx.x >> 5;
    const int tn2  = (N + 31) >> 5;
    const int tm2  = (M + 31) >> 5;
    const int tile = blockIdx.x * (blockDim.x >> 5) + wave;
    if (tile >= tm2 * tn2) return;   // wave-uniform: EXEC stays all-ones for WMMA

    const int tM = tile / tn2;
    const int tN = tile % tn2;
    const int r  = lane & 15;
    const int kb = (lane >> 4) << 3;          // 0 or 8

    int a0 = tM * 32 + r;       a0 = a0 < M ? a0 : M - 1;
    int a1 = tM * 32 + 16 + r;  a1 = a1 < M ? a1 : M - 1;
    int b0 = tN * 32 + r;       b0 = b0 < N ? b0 : N - 1;
    int b1 = tN * 32 + 16 + r;  b1 = b1 < N ? b1 : N - 1;
    const float* Ap0 = A + (size_t)a0 * K + kb;
    const float* Ap1 = A + (size_t)a1 * K + kb;
    const float* Wp0 = W + (size_t)b0 * K + kb;
    const float* Wp1 = W + (size_t)b1 * K + kb;

    v8f c00 = {}, c01 = {}, c10 = {}, c11 = {};
    for (int k0 = 0; k0 < K; k0 += 32) {
        if (k0 + 32 < K) {                      // uniform condition
            __builtin_prefetch(Ap0 + k0 + 32, 0, 1);
            __builtin_prefetch(Ap1 + k0 + 32, 0, 1);
            __builtin_prefetch(Wp0 + k0 + 32, 0, 1);
            __builtin_prefetch(Wp1 + k0 + 32, 0, 1);
        }
        v16h fa0 = frag_from(Ap0 + k0);
        v16h fa1 = frag_from(Ap1 + k0);
        v16h fb0 = frag_from(Wp0 + k0);
        v16h fb1 = frag_from(Wp1 + k0);
        c00 = __builtin_amdgcn_wmma_f32_16x16x32_f16(false, fa0, false, fb0, (short)0, c00, false, false);
        c01 = __builtin_amdgcn_wmma_f32_16x16x32_f16(false, fa0, false, fb1, (short)0, c01, false, false);
        c10 = __builtin_amdgcn_wmma_f32_16x16x32_f16(false, fa1, false, fb0, (short)0, c10, false, false);
        c11 = __builtin_amdgcn_wmma_f32_16x16x32_f16(false, fa1, false, fb1, (short)0, c11, false, false);
    }
    // C/D layout: VGPR i -> M = i + (lane>=16 ? 8 : 0), N = lane&15
    const int n0  = tN * 32 + (lane & 15);
    const int n1  = n0 + 16;
    const int mb0 = tM * 32 + ((lane >> 4) << 3);
    const int mb1 = mb0 + 16;
    #pragma unroll
    for (int i = 0; i < 8; ++i) {
        int m0 = mb0 + i, m1 = mb1 + i;
        if (m0 < M && n0 < N) C[(size_t)m0 * N + n0] = c00[i];
        if (m0 < M && n1 < N) C[(size_t)m0 * N + n1] = c01[i];
        if (m1 < M && n0 < N) C[(size_t)m1 * N + n0] = c10[i];
        if (m1 < M && n1 < N) C[(size_t)m1 * N + n1] = c11[i];
    }
}

// ---------------------------------------------------------------------------
// Depthwise causal conv1d (k=4, left pad 3) over xc = xz[:, :, 0:Di] + bias, SiLU.
// xz: [B*L, 2*Di] ; u: [B*L, Di]
// ---------------------------------------------------------------------------
__global__ void dwconv_silu_kernel(const float* __restrict__ xz, const float* __restrict__ wconv,
                                   const float* __restrict__ cb, float* __restrict__ u,
                                   int L, int Di) {
    int idx = blockIdx.x * blockDim.x + threadIdx.x;
    int total = 4 * L * Di;
    if (idx >= total) return;
    int d = idx % Di; int t2 = idx / Di;
    int t = t2 % L;   int b = t2 / L;
    float acc = cb[d];
    const float* wp = wconv + d * 4;
    #pragma unroll
    for (int j = 0; j < 4; ++j) {
        int tt = t - 3 + j;
        if (tt >= 0) acc += wp[j] * xz[((size_t)(b * L + tt)) * (2 * Di) + d];
    }
    u[idx] = acc / (1.f + __expf(-acc));
}

// ---------------------------------------------------------------------------
// dt = softplus(xdb[:, :R] @ dt_w^T + dt_b)   (tiny K <= 32: plain VALU)
// ---------------------------------------------------------------------------
__global__ void dt_kernel(const float* __restrict__ xdb, const float* __restrict__ dt_w,
                          const float* __restrict__ dt_b, float* __restrict__ dt,
                          int BL, int Di, int R) {
    int idx = blockIdx.x * blockDim.x + threadIdx.x;
    int total = BL * Di;
    if (idx >= total) return;
    int d = idx % Di; int bt = idx / Di;
    const float* xr = xdb + (size_t)bt * (R + 64);
    const float* wr = dt_w + (size_t)d * R;
    float acc = dt_b[d];
    for (int r = 0; r < R; ++r) acc += xr[r] * wr[r];
    dt[idx] = acc > 20.f ? acc : log1pf(__expf(acc));
}

// ---------------------------------------------------------------------------
// Selective scan, fused with output gating y *= silu(z).
// One block per batch element; one lane per channel d; h[32],A[32] in VGPRs.
// B_t/C_t staged through LDS (shared by all channels).
// y written in-place over u.
// ---------------------------------------------------------------------------
__global__ void scan_kernel(float* __restrict__ u, const float* __restrict__ dt,
                            const float* __restrict__ xdb, const float* __restrict__ A_log,
                            const float* __restrict__ Dp, const float* __restrict__ xz,
                            int L, int Di, int R) {
    const int b = blockIdx.x;
    const int d = threadIdx.x;
    __shared__ float BC[64];
    float A[32], h[32];
    #pragma unroll
    for (int n = 0; n < 32; ++n) {
        A[n] = -__expf(A_log[(size_t)d * 32 + n]);
        h[n] = 0.f;
    }
    const float Dd = Dp[d];
    const int xs = R + 64;
    for (int t = 0; t < L; ++t) {
        const size_t bt = (size_t)b * L + t;
        if (threadIdx.x < 64) BC[threadIdx.x] = xdb[bt * xs + R + threadIdx.x];
        __syncthreads();
        float dtv = dt[bt * Di + d];
        float uv  = u[bt * Di + d];
        float yv  = 0.f;
        #pragma unroll
        for (int n = 0; n < 32; ++n) {
            float dA = __expf(dtv * A[n]);
            h[n] = dA * h[n] + dtv * BC[n] * uv;
            yv  += h[n] * BC[32 + n];
        }
        yv += uv * Dd;
        float z = xz[bt * (size_t)(2 * Di) + Di + d];
        u[bt * Di + d] = yv * (z / (1.f + __expf(-z)));
        __syncthreads();
    }
}

// ---------------------------------------------------------------------------
// Host orchestration
// ---------------------------------------------------------------------------
static inline void launch_gemm(const float* A, const float* W, float* C,
                               int M, int N, int K, hipStream_t s) {
    int tiles  = ((M + 31) / 32) * ((N + 31) / 32);
    int blocks = (tiles + 7) / 8;                 // 8 waves (256 threads) per block
    wmma_gemm_kernel<<<blocks, 256, 0, s>>>(A, W, C, M, N, K);
}

struct StageBufs {
    float *conv, *seq, *xz, *u, *xdb, *dt;
};

static void run_stage(const float* in, int Cin, int Hin, int Win, int stride, int Cout,
                      void* const* d_in, int base, const StageBufs& wb, float* stage_out,
                      hipStream_t s) {
    auto F = [&](int i) { return (const float*)d_in[i]; };
    const int Ho = (Hin - 1) / stride + 1;
    const int Wo = (Win - 1) / stride + 1;
    const int L  = Ho * Wo;
    const int BL = 4 * L;
    const int Di = 2 * Cout;
    const int R  = Cout / 16;       // dt_rank = ceil(d_model/16), exact here

    {
        int total = 4 * Cout * Ho * Wo;
        conv3x3_bn_relu_kernel<<<(total + 255) / 256, 256, 0, s>>>(
            in, F(base + 0), F(base + 1), F(base + 2), F(base + 3), F(base + 4),
            wb.conv, Cin, Hin, Win, Cout, Ho, Wo, stride);
    }
    {
        int total = 4 * Cout * L;
        chw_to_lc_kernel<<<(total + 255) / 256, 256, 0, s>>>(wb.conv, wb.seq, Cout, L);
    }
    // xz = seq @ in_proj^T          [BL, 2Di]
    launch_gemm(wb.seq, F(base + 5), wb.xz, BL, 2 * Di, Cout, s);
    // depthwise causal conv + SiLU -> u
    {
        int total = BL * Di;
        dwconv_silu_kernel<<<(total + 255) / 256, 256, 0, s>>>(
            wb.xz, F(base + 6), F(base + 7), wb.u, L, Di);
    }
    // xdb = u @ x_proj^T            [BL, R+64]
    launch_gemm(wb.u, F(base + 8), wb.xdb, BL, R + 64, Di, s);
    {
        int total = BL * Di;
        dt_kernel<<<(total + 255) / 256, 256, 0, s>>>(
            wb.xdb, F(base + 9), F(base + 10), wb.dt, BL, Di, R);
    }
    // selective scan + gate, y in-place over u
    scan_kernel<<<4, Di, 0, s>>>(wb.u, wb.dt, wb.xdb, F(base + 11), F(base + 12),
                                 wb.xz, L, Di, R);
    // out_seq = y @ out_proj^T      [BL, Cout]   (reuse seq buffer)
    launch_gemm(wb.u, F(base + 13), wb.seq, BL, Cout, Di, s);
    {
        int total = 4 * Cout * L;
        lc_to_chw_kernel<<<(total + 255) / 256, 256, 0, s>>>(wb.seq, stage_out, Cout, L);
    }
}

extern "C" void kernel_launch(void* const* d_in, const int* in_sizes, int n_in,
                              void* d_out, int out_size, void* d_ws, size_t ws_size,
                              hipStream_t stream) {
    (void)in_sizes; (void)n_in; (void)out_size; (void)ws_size;
    const float* x = (const float*)d_in[0];
    auto F = [&](int i) { return (const float*)d_in[i]; };
    float* ws  = (float*)d_ws;
    float* out = (float*)d_out;

    // workspace layout (float offsets); stem region reused as xz, y in-place in u,
    // out_seq reuses seq.  Total ~9.69M floats (~39 MB).
    float* stem = ws + 0;          // 3,211,264
    float* pool = ws + 3211264;    //   802,816
    StageBufs wb;
    wb.conv = ws + 4014080;        //   802,816
    wb.seq  = ws + 4816896;        //   802,816
    wb.xz   = ws + 0;              // 3,211,264 (reuse stem)
    wb.u    = ws + 5619712;        // 1,605,632
    wb.xdb  = ws + 7225344;        //   852,992
    wb.dt   = ws + 8078336;        // 1,605,632

    {
        int total = 4 * 64 * 112 * 112;
        stem_conv_kernel<<<(total + 255) / 256, 256, 0, stream>>>(
            x, F(1), F(2), F(3), F(4), F(5), F(6), stem);
    }
    {
        int total = 4 * 64 * 56 * 56;
        maxpool_kernel<<<(total + 255) / 256, 256, 0, stream>>>(stem, pool);
    }

    float* c1 = out + 0;
    float* c2 = out + 802816;
    float* c3 = out + 1204224;
    float* c4 = out + 1404928;

    run_stage(pool, 64,  56, 56, 1,  64, d_in,  7, wb, c1, stream);
    run_stage(c1,   64,  56, 56, 2, 128, d_in, 21, wb, c2, stream);
    run_stage(c2,  128,  28, 28, 2, 256, d_in, 35, wb, c3, stream);
    run_stage(c3,  256,  14, 14, 2, 512, d_in, 49, wb, c4, stream);
}
